// CrossModalAttention_9844065042451
// MI455X (gfx1250) — matrix operand order repaired and checked by reading, not compiled, not run on hardware
//
#include <hip/hip_runtime.h>

// ---------------------------------------------------------------------------
// CrossModalAttention for MI455X (gfx1250, wave32, WMMA bf16 w/ f32 accum)
//
// Pipeline (all bf16 data produced in workspace):
//   1) weights f32 [K x N] -> bf16 transposed [N x K]      (contiguous B-frags)
//   2) query / key_value f32 -> bf16
//   3) GEMM: Q = q @ Wq + bq          -> bf16 [B,LQ,H]
//            K = kv @ Wk + bk         -> bf16 [B,LKV,H]
//            V = kv @ Wv + bv         -> bf16 transposed [B,H,LKV]
//   4) flash attention  -> context bf16 [B,LQ,H]
//   5) GEMM: out = ctx @ Wo + bo      -> f32 d_out
//
// v3b: async global->LDS staging (GLOBAL_LOAD_ASYNC_TO_LDS_B128 + ASYNCcnt)
//      with corrected builtin pointer types (v4i32* AS1 / AS3); load+ds_store
//      fallback; LDS double buffering; register double-buffering; parallel
//      softmax; global_prefetch_b8 on streaming operands.
// ---------------------------------------------------------------------------

#define B_SZ   8
#define LQ_SZ  1024
#define LKV_SZ 2048
#define H_SZ   1024

typedef __attribute__((ext_vector_type(8)))  __bf16 v8bf_t;
typedef __attribute__((ext_vector_type(16))) __bf16 v16bf_t;
typedef __attribute__((ext_vector_type(8)))  float  v8f_t;
typedef __attribute__((ext_vector_type(4)))  int    v4i_t;

#if defined(__AMDGCN__) && __has_builtin(__builtin_amdgcn_global_load_async_to_lds_b128)
#define HAVE_ASYNC_LDS 1
#else
#define HAVE_ASYNC_LDS 0
#endif

#define AS_GLOBAL __attribute__((address_space(1)))
#define AS_LOCAL  __attribute__((address_space(3)))

__device__ __forceinline__ void async_lds_b128(const void* gptr, void* lptr) {
#if HAVE_ASYNC_LDS
    __builtin_amdgcn_global_load_async_to_lds_b128(
        (AS_GLOBAL v4i_t*)gptr, (AS_LOCAL v4i_t*)lptr, 0, 0);
#else
    *(v8bf_t*)lptr = *(const v8bf_t*)gptr;
#endif
}

__device__ __forceinline__ void async_wait0() {
#if HAVE_ASYNC_LDS
#if __has_builtin(__builtin_amdgcn_s_wait_asynccnt)
    __builtin_amdgcn_s_wait_asynccnt(0);
#else
    asm volatile("s_wait_asynccnt 0x0" ::: "memory");
#endif
#endif
}

__device__ __forceinline__ __bf16 f2bf(float f) {
    unsigned u = __builtin_bit_cast(unsigned, f);
    unsigned r = (u + 0x7FFFu + ((u >> 16) & 1u)) >> 16;
    unsigned short s = (unsigned short)r;
    return __builtin_bit_cast(__bf16, s);
}

__device__ __forceinline__ v16bf_t frag_cat(v8bf_t lo, v8bf_t hi) {
    return __builtin_shufflevector(lo, hi, 0,1,2,3,4,5,6,7,8,9,10,11,12,13,14,15);
}

// A-matrix 16x32 bf16 fragment (ISA 7.12.2): lane holds row (lane&15);
// elements 0..7 = K[klo..klo+7], elements 8..15 = K[16+klo..16+klo+7],
// klo = (lane<16) ? 0 : 8.  rowbase points at A[row]+k0 (k0 % 32 == 0).
__device__ __forceinline__ v16bf_t load_a_frag(const __bf16* rowbase, int lane) {
    int klo = (lane < 16) ? 0 : 8;
    v8bf_t lo = *(const v8bf_t*)(rowbase + klo);
    v8bf_t hi = *(const v8bf_t*)(rowbase + 16 + klo);
    return frag_cat(lo, hi);
}

// B-matrix 32x16 bf16 fragment: lane holds column (lane&15);
// elements 0..7 = K[klo..klo+7], elements 8..15 = K[16+klo..16+klo+7], klo = (lane<16) ? 0 : 8.
__device__ __forceinline__ v16bf_t load_b_frag(const __bf16* colbase, int lane) {
    int koff = (lane < 16) ? 0 : 8;
    v8bf_t lo = *(const v8bf_t*)(colbase + koff);
    v8bf_t hi = *(const v8bf_t*)(colbase + 16 + koff);
    return frag_cat(lo, hi);
}

__device__ __forceinline__ v8f_t wmma_bf16(v16bf_t a, v16bf_t b, v8f_t c) {
    // D = A x B + C, f32 accumulate; emits v_wmma_f32_16x16x32_bf16
    return __builtin_amdgcn_wmma_f32_16x16x32_bf16(false, a, false, b,
                                                   (short)0, c, false, false);
}

// ---------------------------------------------------------------------------
// Conversion kernels
// ---------------------------------------------------------------------------

// W f32 [K x N] (K=N=1024) -> Wt bf16 [N x K]
__global__ __launch_bounds__(256) void cvt_transpose_w(const float* __restrict__ W,
                                                       __bf16* __restrict__ Wt) {
    int idx = blockIdx.x * 256 + threadIdx.x;          // 1M elements
    int n = idx >> 10;
    int k = idx & 1023;
    Wt[(size_t)n * 1024 + k] = f2bf(W[(size_t)k * 1024 + n]);
}

__global__ __launch_bounds__(256) void cvt_f32_bf16(const float* __restrict__ in,
                                                    __bf16* __restrict__ out,
                                                    int count) {
    int idx = blockIdx.x * 256 + threadIdx.x;
    if (idx < count) out[idx] = f2bf(in[idx]);
}

// ---------------------------------------------------------------------------
// bf16 GEMM: out[M x N] = A[M x K] @ Bt[N x K]^T + bias
//   mode 0: bf16 row-major [M x N]
//   mode 1: bf16 transposed per batch (V -> Vt[B,H,LKV])
//   mode 2: f32 row-major [M x N]
// Block = 256 threads (8 waves); wave w owns rows [blk.x*128 + w*16, +16),
// cols [blk.y*64, +64).  B tile (64 cols x 32 K) staged in LDS via async
// global->LDS copies, double-buffered; A fragment double-buffered in regs.
// ---------------------------------------------------------------------------
__global__ __launch_bounds__(256) void gemm_bf16_kernel(const __bf16* __restrict__ A,
                                                        const __bf16* __restrict__ Bt,
                                                        const float* __restrict__ bias,
                                                        void* __restrict__ outp,
                                                        int M, int N, int K,
                                                        int mode, int lkv) {
    __shared__ __align__(16) __bf16 Bs[2][64 * 32];    // 2 x 4 KB

    int tid  = threadIdx.x;
    int wave = tid >> 5;
    int lane = tid & 31;

    int m0 = blockIdx.x * 128 + wave * 16;
    int n0 = blockIdx.y * 64;

    const __bf16* arow = A + (size_t)(m0 + (lane & 15)) * K;

    // Cooperative B-tile stage: 64 cols x 32 K, one 16-byte chunk per thread,
    // issued as async global->LDS (ASYNCcnt) when available.
    auto stageB = [&](int buf, int k0) {
        int col  = tid >> 2;
        int part = tid & 3;
        async_lds_b128(Bt + (size_t)(n0 + col) * K + k0 + part * 8,
                       &Bs[buf][col * 32 + part * 8]);
    };

    v8f_t acc[4] = {};

    stageB(0, 0);
    v16bf_t a_cur = load_a_frag(arow, lane);
    async_wait0();
    __syncthreads();

    int nsteps = K / 32;
    for (int ks = 0; ks < nsteps; ++ks) {
        int k0   = ks * 32;
        int cbuf = ks & 1;

        v16bf_t a_nxt = a_cur;
        if (ks + 1 < nsteps) {
            stageB(cbuf ^ 1, k0 + 32);
            a_nxt = load_a_frag(arow + k0 + 32, lane);
        }
        if ((ks & 7) == 0)
            __builtin_prefetch(arow + k0 + 512, 0, 1);   // global_prefetch_b8

#pragma unroll
        for (int t = 0; t < 4; ++t) {
            const __bf16* cb = &Bs[cbuf][(t * 16 + (lane & 15)) * 32];
            v16bf_t b = load_b_frag(cb, lane);
            acc[t] = wmma_bf16(a_cur, b, acc[t]);
        }
        a_cur = a_nxt;
        async_wait0();          // next buffer resident before the barrier
        __syncthreads();
    }

    int lh = (lane < 16) ? 0 : 8;
#pragma unroll
    for (int t = 0; t < 4; ++t) {
        int n = n0 + t * 16 + (lane & 15);
        float bv = bias[n];
#pragma unroll
        for (int r = 0; r < 8; ++r) {
            int m = m0 + r + lh;
            float v = acc[t][r] + bv;
            if (mode == 2) {
                ((float*)outp)[(size_t)m * N + n] = v;
            } else if (mode == 0) {
                ((__bf16*)outp)[(size_t)m * N + n] = f2bf(v);
            } else {
                int bb = m / lkv;
                int k  = m - bb * lkv;
                ((__bf16*)outp)[(size_t)bb * N * lkv + (size_t)n * lkv + k] = f2bf(v);
            }
        }
    }
}

// ---------------------------------------------------------------------------
// Flash attention: one block (8 waves) per (batch, 16-row Q tile).
//   Q bf16 [B,LQ,H], K bf16 [B,LKV,H], Vt bf16 [B,H,LKV],
//   mask u8 [B,LKV] -> ctx bf16 [B,LQ,H]
// ---------------------------------------------------------------------------
__global__ __launch_bounds__(256) void attn_kernel(const __bf16* __restrict__ Qb,
                                                   const __bf16* __restrict__ Kb,
                                                   const __bf16* __restrict__ Vt,
                                                   const unsigned char* __restrict__ mask,
                                                   __bf16* __restrict__ ctx) {
    __shared__ __align__(16) __bf16 Qs[16 * H_SZ];     // 32 KB
    __shared__ __align__(16) float  Ss[16 * 128];      //  8 KB
    __shared__ __align__(16) __bf16 Ps[16 * 128];      //  4 KB
    __shared__ float red[256];                         //  1 KB
    __shared__ float sm_m[16], sm_l[16], sm_alpha[16];

    int tid  = threadIdx.x;
    int wave = tid >> 5;
    int lane = tid & 31;
    int lh   = (lane < 16) ? 0 : 8;

    int b  = blockIdx.x >> 6;          // LQ/16 = 64 tiles per batch
    int q0 = (blockIdx.x & 63) << 4;

    // Stage the 16 x H Q tile into LDS via async global->LDS copies.
    {
        const __bf16* src = Qb + ((size_t)b * LQ_SZ + q0) * H_SZ;
        for (int i = tid; i < 16 * H_SZ / 8; i += 256)
            async_lds_b128(src + i * 8, Qs + i * 8);
        async_wait0();
    }
    if (tid < 16) { sm_m[tid] = -3.0e38f; sm_l[tid] = 0.0f; }

    v8f_t oacc[8] = {};
    __syncthreads();

    for (int kv0 = 0; kv0 < LKV_SZ; kv0 += 128) {
        // ---- scores: S[16 x 16] for this wave's 16 KV columns --------------
        int ncol = kv0 + wave * 16 + (lane & 15);
        const __bf16* krow = Kb + ((size_t)b * LKV_SZ + ncol) * H_SZ;
        const __bf16* qrow = Qs + (lane & 15) * H_SZ;

        v8f_t s = {};
        v16bf_t qa = load_a_frag(qrow, lane);
        v16bf_t kb = load_b_frag(krow, lane);
        for (int h0 = 0; h0 < H_SZ; h0 += 32) {
            v16bf_t qa_n = qa, kb_n = kb;
            if (h0 + 32 < H_SZ) {
                qa_n = load_a_frag(qrow + h0 + 32, lane);
                kb_n = load_b_frag(krow + h0 + 32, lane);
            }
            if ((h0 & 255) == 0)
                __builtin_prefetch(krow + h0 + 256, 0, 1);
            s  = wmma_bf16(qa, kb, s);
            qa = qa_n;
            kb = kb_n;
        }

        bool mv = mask[(size_t)b * LKV_SZ + ncol] != 0;
        int col = wave * 16 + (lane & 15);
#pragma unroll
        for (int r = 0; r < 8; ++r) {
            float v = s[r] * 0.03125f;                 // 1/sqrt(1024)
            Ss[(r + lh) * 128 + col] = mv ? v : -1.0e9f;
        }
        __syncthreads();

        // ---- online softmax, 16 threads per row, 8 cols per thread ---------
        {
            int row = tid >> 4, sub = tid & 15;
            const float* srow = Ss + row * 128 + sub * 8;
            float mold = sm_m[row];

            float lmax = -3.0e38f;
#pragma unroll
            for (int j = 0; j < 8; ++j) lmax = fmaxf(lmax, srow[j]);
            red[tid] = lmax;
            __syncthreads();

            float tmax = -3.0e38f;
#pragma unroll
            for (int j = 0; j < 16; ++j) tmax = fmaxf(tmax, red[(row << 4) + j]);
            float mnew = fmaxf(mold, tmax);

            __bf16* prow = Ps + row * 128 + sub * 8;
            float sum = 0.0f;
#pragma unroll
            for (int j = 0; j < 8; ++j) {
                float p = __expf(srow[j] - mnew);
                sum += p;
                prow[j] = f2bf(p);
            }
            red[tid] = sum;
            __syncthreads();

            if (sub == 0) {
                float tot = 0.0f;
#pragma unroll
                for (int j = 0; j < 16; ++j) tot += red[(row << 4) + j];
                float alpha = __expf(mold - mnew);
                sm_l[row]     = sm_l[row] * alpha + tot;
                sm_m[row]     = mnew;
                sm_alpha[row] = alpha;
            }
        }
        __syncthreads();

        // ---- O = O*alpha + P @ V_slice  (this wave: HID [wave*128,+128)) ---
        float al[8];
#pragma unroll
        for (int r = 0; r < 8; ++r) al[r] = sm_alpha[r + lh];

        v16bf_t pa[4];
#pragma unroll
        for (int kk = 0; kk < 4; ++kk)
            pa[kk] = load_a_frag(Ps + (lane & 15) * 128 + kk * 32, lane);

        const __bf16* vbase = Vt + (size_t)b * H_SZ * LKV_SZ;
        int hbase = wave * 128;

        v16bf_t vb_cur[4];
        {
            const __bf16* vc0 =
                vbase + (size_t)(hbase + (lane & 15)) * LKV_SZ + kv0;
#pragma unroll
            for (int kk = 0; kk < 4; ++kk)
                vb_cur[kk] = load_b_frag(vc0 + kk * 32, lane);
        }
#pragma unroll
        for (int t = 0; t < 8; ++t) {
            v16bf_t vb_nxt[4];
            if (t < 7) {
                const __bf16* vcn =
                    vbase + (size_t)(hbase + (t + 1) * 16 + (lane & 15)) * LKV_SZ + kv0;
#pragma unroll
                for (int kk = 0; kk < 4; ++kk)
                    vb_nxt[kk] = load_b_frag(vcn + kk * 32, lane);
            } else {
#pragma unroll
                for (int kk = 0; kk < 4; ++kk) vb_nxt[kk] = vb_cur[kk];
            }

            v8f_t o = oacc[t];
#pragma unroll
            for (int r = 0; r < 8; ++r) o[r] *= al[r];
#pragma unroll
            for (int kk = 0; kk < 4; ++kk)
                o = wmma_bf16(pa[kk], vb_cur[kk], o);
            oacc[t] = o;
#pragma unroll
            for (int kk = 0; kk < 4; ++kk) vb_cur[kk] = vb_nxt[kk];
        }
        __syncthreads();
    }

    // ---- normalize by l and store bf16 context -----------------------------
    float rl[8];
#pragma unroll
    for (int r = 0; r < 8; ++r) rl[r] = 1.0f / sm_l[r + lh];
#pragma unroll
    for (int t = 0; t < 8; ++t) {
        int h = wave * 128 + t * 16 + (lane & 15);
#pragma unroll
        for (int r = 0; r < 8; ++r) {
            int m = q0 + r + lh;
            ctx[((size_t)b * LQ_SZ + m) * H_SZ + h] = f2bf(oacc[t][r] * rl[r]);
        }
    }
}

// ---------------------------------------------------------------------------
// Host launcher
// ---------------------------------------------------------------------------
extern "C" void kernel_launch(void* const* d_in, const int* in_sizes, int n_in,
                              void* d_out, int out_size, void* d_ws, size_t ws_size,
                              hipStream_t stream) {
    (void)in_sizes; (void)n_in; (void)out_size; (void)ws_size;

    const float* query     = (const float*)d_in[0];
    const float* key_value = (const float*)d_in[1];
    const unsigned char* key_mask = (const unsigned char*)d_in[2]; // bool8
    const float* Wq = (const float*)d_in[3];
    const float* bq = (const float*)d_in[4];
    const float* Wk = (const float*)d_in[5];
    const float* bk = (const float*)d_in[6];
    const float* Wv = (const float*)d_in[7];
    const float* bv = (const float*)d_in[8];
    const float* Wo = (const float*)d_in[9];
    const float* bo = (const float*)d_in[10];
    float* out = (float*)d_out;

    const int H = H_SZ, LQ = LQ_SZ, LKV = LKV_SZ, Bn = B_SZ;
    const int Mq  = Bn * LQ;    // 8192
    const int Mkv = Bn * LKV;   // 16384

    char* ws = (char*)d_ws;
    size_t off = 0;
    auto take = [&](size_t bytes) -> void* {
        void* p = ws + off;
        off += (bytes + 255) & ~(size_t)255;
        return p;
    };
    __bf16* WqT  = (__bf16*)take((size_t)H * H * 2);
    __bf16* WkT  = (__bf16*)take((size_t)H * H * 2);
    __bf16* WvT  = (__bf16*)take((size_t)H * H * 2);
    __bf16* WoT  = (__bf16*)take((size_t)H * H * 2);
    __bf16* qbf  = (__bf16*)take((size_t)Mq  * H * 2);
    __bf16* kvbf = (__bf16*)take((size_t)Mkv * H * 2);
    __bf16* Qbf  = (__bf16*)take((size_t)Mq  * H * 2);
    __bf16* Kbf  = (__bf16*)take((size_t)Mkv * H * 2);
    __bf16* Vtbf = (__bf16*)take((size_t)Mkv * H * 2);   // [B,H,LKV]
    __bf16* ctxb = (__bf16*)take((size_t)Mq  * H * 2);

    // 1) weights -> bf16 transposed
    cvt_transpose_w<<<(H * H) / 256, 256, 0, stream>>>(Wq, WqT);
    cvt_transpose_w<<<(H * H) / 256, 256, 0, stream>>>(Wk, WkT);
    cvt_transpose_w<<<(H * H) / 256, 256, 0, stream>>>(Wv, WvT);
    cvt_transpose_w<<<(H * H) / 256, 256, 0, stream>>>(Wo, WoT);

    // 2) activations -> bf16
    cvt_f32_bf16<<<(Mq * H) / 256, 256, 0, stream>>>(query, qbf, Mq * H);
    cvt_f32_bf16<<<(Mkv * H) / 256, 256, 0, stream>>>(key_value, kvbf, Mkv * H);

    // 3) projections
    dim3 gq(Mq / 128, H / 64), gkv(Mkv / 128, H / 64);
    gemm_bf16_kernel<<<gq,  256, 0, stream>>>(qbf,  WqT, bq, Qbf,  Mq,  H, H, 0, LKV);
    gemm_bf16_kernel<<<gkv, 256, 0, stream>>>(kvbf, WkT, bk, Kbf,  Mkv, H, H, 0, LKV);
    gemm_bf16_kernel<<<gkv, 256, 0, stream>>>(kvbf, WvT, bv, Vtbf, Mkv, H, H, 1, LKV);

    // 4) attention
    attn_kernel<<<Bn * (LQ / 16), 256, 0, stream>>>(Qbf, Kbf, Vtbf, key_mask, ctxb);

    // 5) output projection (f32 out)
    gemm_bf16_kernel<<<gq, 256, 0, stream>>>(ctxb, WoT, bo, out, Mq, H, H, 2, LKV);
}
